// Shift_32762010534460
// MI455X (gfx1250) — matrix operand order repaired and checked by reading, not compiled, not run on hardware
//
#include <hip/hip_runtime.h>

// CDNA5 / gfx1250: wave32, WMMA 16x16x32 bf16 (f32 accumulate), TDM weight DMA.
typedef __attribute__((ext_vector_type(16))) __bf16 v16bf;
typedef __attribute__((ext_vector_type(8)))  __bf16 v8bf;
typedef __attribute__((ext_vector_type(8)))  float  v8f;
typedef __attribute__((ext_vector_type(4)))  unsigned int v4u;
typedef __attribute__((ext_vector_type(8)))  int v8i;
typedef __attribute__((ext_vector_type(4)))  int v4i;

#define CIN    64
#define COUT   64
#define HH     56
#define WW2    56
#define HWSZ   3136                 // 56*56
#define KTOT   576                  // 9 taps * 64 channels (tap-major)
#define NSEG   8
#define ROWS_PER_WG 4
#define TROWS  (ROWS_PER_WG + 2)    // 6 rows incl. halo
#define TCOLS  (WW2 + 2)            // 58 cols incl. halo
#define NTHREADS 256                // 8 waves of 32

#define WLDS_BYTES (COUT * KTOT * 2)           // 73,728 (bf16 weights)
#define XLDS_BYTES (TROWS * TCOLS * CIN * 2)   // 44,544 (bf16 shifted tile)
#define DTS_BYTES  (CIN * 4)
#define SMEM_BYTES (WLDS_BYTES + XLDS_BYTES + DTS_BYTES)   // 118,528 < 320KB/WGP

// ---------------- prep: permute weights K:(ch*9+tap) -> (tap*64+ch), fp32->bf16 ----
__global__ __launch_bounds__(256)
void tsm_prep_weights(const float* __restrict__ net_w, __bf16* __restrict__ wperm) {
    const int i = blockIdx.x * 256 + threadIdx.x;       // 0..36863, coalesced read
    const int cout = i / KTOT;
    const int rem  = i - cout * KTOT;
    const int ch   = rem / 9;
    const int tap  = rem - ch * 9;
    wperm[cout * KTOT + tap * CIN + ch] = (__bf16)net_w[i];
}

// ---------------- main: TSM shift + implicit-GEMM 3x3 conv via WMMA ----------------
__global__ __launch_bounds__(NTHREADS)
void tsm_conv3x3_wmma_bf16(const float* __restrict__ x,       // (128,64,56,56)
                           const float* __restrict__ ta_w,    // (64,1,3)
                           const __bf16* __restrict__ wperm,  // (64,576) bf16, tap-major
                           float* __restrict__ out)           // (128,64,56,56)
{
    extern __shared__ __attribute__((aligned(32))) char smem[];
    __bf16* wlds = (__bf16*)smem;                        // [cout][tap*64+ch]
    __bf16* xlds = (__bf16*)(smem + WLDS_BYTES);         // [trow][tcol][ch]
    int*    dts  = (int*)(smem + WLDS_BYTES + XLDS_BYTES);

    const int tid  = threadIdx.x;
    const int lane = tid & 31;
    const int wave = tid >> 5;

    const int n     = blockIdx.x / (HH / ROWS_PER_WG);   // image
    const int strip = blockIdx.x - n * (HH / ROWS_PER_WG);
    const int oh0   = strip * ROWS_PER_WG;               // first output row
    const int t     = n & (NSEG - 1);                    // frame within segment

    // --- 1) per-channel temporal shift direction (ta_weight rows are one-hot) ---
    if (tid < CIN) {
        const float w0 = ta_w[tid * 3 + 0];              // take x[t-1]
        const float w2 = ta_w[tid * 3 + 2];              // take x[t+1]
        dts[tid] = (w2 != 0.0f) ? 1 : ((w0 != 0.0f) ? -1 : 0);
    }

    // --- 2) kick off the weight DMA (73,728 B contiguous -> LDS) on wave 0 via the
    //        Tensor Data Mover; overlaps with zero-fill and x-tile staging below.
#if __has_builtin(__builtin_amdgcn_tensor_load_to_lds)
    if (wave == 0) {
        const unsigned lds_addr = (unsigned)(uintptr_t)(void*)wlds;   // LDS aperture: addr[31:0]
        const unsigned long long ga = (unsigned long long)(uintptr_t)(const void*)wperm;
        const unsigned NW = WLDS_BYTES / 4;              // 18,432 dwords
        v4u g0;                                          // D# group 0 (ISA 8.3)
        g0[0] = 1u;                                      // count=1, user flags 0
        g0[1] = lds_addr;                                // lds_addr[31:0]
        g0[2] = (unsigned)(ga & 0xFFFFFFFFu);            // global_addr[31:0]
        g0[3] = (unsigned)((ga >> 32) & 0x1FFFFFFu)      // global_addr[56:32]
              | (2u << 30);                              // type=2 ("image")
        v8i g1;                                          // D# group 1 (ISA 8.4)
        g1[0] = (int)(2u << 16);                         // data_size=4B; mask/flags 0
        g1[1] = (int)(NW << 16);                         // tensor_dim0[15:0]  (=0x4800)
        g1[2] = (int)(1u << 16);                         // tensor_dim0 hi=0; tensor_dim1=1
        g1[3] = (int)((NW & 0xFFFFu) << 16);             // tensor_dim1 hi=0; tile_dim0=NW
        g1[4] = 1;                                       // tile_dim1=1; tile_dim2=0
        g1[5] = (int)NW;                                 // tensor_dim0_stride[31:0]
        g1[6] = (int)((NW & 0xFFFFu) << 16);             // stride0 hi=0; stride1 lo=NW
        g1[7] = 0;                                       // tensor_dim1_stride hi
        v4i g2 = {0, 0, 0, 0};                           // 2-D tensor: groups 2/3 unused
        v4i g3 = {0, 0, 0, 0};
        v8i g4 = {0, 0, 0, 0, 0, 0, 0, 0};               // 6-arg toolchain: extra group, zero
        __builtin_amdgcn_tensor_load_to_lds(g0, g1, g2, g3, g4, 0);
    }
#else
    #pragma unroll 4
    for (int i = tid; i < WLDS_BYTES / 16; i += NTHREADS)
        ((uint4*)wlds)[i] = ((const uint4*)wperm)[i];
#endif

    // --- 3) zero the input tile (halo stays zero => no bounds checks in GEMM) ---
    {
        const uint4 z = make_uint4(0u, 0u, 0u, 0u);
        uint4* p = (uint4*)xlds;
        #pragma unroll 4
        for (int i = tid; i < XLDS_BYTES / 16; i += NTHREADS) p[i] = z;
    }
    __syncthreads();  // dts + zero-fill visible before x staging

    // --- 4) stage temporally-shifted input tile: xlds[r][c][ch], bf16 ---
    const int NIT = TROWS * CIN * (WW2 / 4);             // 5376 float4 chunks
    #pragma unroll 2
    for (int i = tid; i < NIT; i += NTHREADS) {
        const int q  = i % (WW2 / 4);
        const int rc = i / (WW2 / 4);
        const int ch = rc % CIN;
        const int r  = rc / CIN;
        const int ih = oh0 - 1 + r;
        const int dt = dts[ch];
        const int ts = t + dt;
        if ((unsigned)ih < (unsigned)HH && (unsigned)ts < (unsigned)NSEG) {
            const float4 v = *(const float4*)(x + ((size_t)(n + dt) * CIN + ch) * HWSZ
                                                + ih * WW2 + q * 4);
            __bf16* dst = &xlds[(r * TCOLS + 1 + q * 4) * CIN + ch];
            dst[0 * CIN] = (__bf16)v.x;
            dst[1 * CIN] = (__bf16)v.y;
            dst[2 * CIN] = (__bf16)v.z;
            dst[3 * CIN] = (__bf16)v.w;
        }
    }
#if __has_builtin(__builtin_amdgcn_tensor_load_to_lds)
    if (wave == 0) __builtin_amdgcn_s_wait_tensorcnt(0); // drain weight DMA
#endif
    __syncthreads();

    // --- 5) GEMM: M=224, N=64, K=576 (tap-major). wave = (ntile<<1)|mhalf. ---
    const int ntile = wave >> 1;
    const int mhalf = wave & 1;
    const int nIdx  = lane & 15;
    const int mbase = (lane < 16) ? 0 : 8;

    int aoff[7];                                         // per-M-tile A base offsets
    #pragma unroll
    for (int mt = 0; mt < 7; ++mt) {
        const int m  = (mhalf * 7 + mt) * 16 + nIdx;     // this lane's A row (pixel)
        const int rm = m / WW2;
        const int cm = m - rm * WW2;
        aoff[mt] = (rm * TCOLS + cm) * CIN * 2;
    }
    const char* xbase = (const char*)xlds;
    const char* wbase = (const char*)&wlds[(ntile * 16 + nIdx) * KTOT];
    // A layout (matches B): lanes 0-15 hold K 0..15; lanes 16-31 hold K 16..31.
    const int cA  = (lane < 16) ? 0 : 16;
    // B layout (matches A): lanes 0-15 hold K 0..15; lanes 16-31 hold K 16..31.
    const int kbB = (lane < 16) ? 0 : 16;

    v8f acc[7] = {};

    #pragma unroll
    for (int tap = 0; tap < 9; ++tap) {
        const int kh   = tap / 3;
        const int kw   = tap - kh * 3;
        const int toff = (kh * TCOLS + kw) * CIN * 2;    // compile-time const
        #pragma unroll
        for (int c0 = 0; c0 < CIN; c0 += 32) {           // two K-slices per tap
            const v16bf bfrag =
                *(const v16bf*)(wbase + (tap * CIN + c0 + kbB) * 2);
            #pragma unroll
            for (int mt = 0; mt < 7; ++mt) {
                const char* ap = xbase + aoff[mt] + toff + (c0 + cA) * 2;
                const v8bf lo = *(const v8bf*)(ap);      // K = c0+cA    .. +7
                const v8bf hi = *(const v8bf*)(ap + 16); // K = c0+cA+8  .. +15
                v16bf afrag;
                #pragma unroll
                for (int j = 0; j < 8; ++j) { afrag[j] = lo[j]; afrag[j + 8] = hi[j]; }
                acc[mt] = __builtin_amdgcn_wmma_f32_16x16x32_bf16(
                    /*neg_a=*/false, afrag, /*neg_b=*/false, bfrag,
                    /*c_mod=*/(short)0, acc[mt], /*reuse_a=*/false, /*reuse_b=*/false);
            }
        }
    }

    // --- 6) epilogue: oh*56+ow == oh0*56 + m; obase+m0 is 16B-aligned -> b128 stores.
    const int cout = ntile * 16 + nIdx;
    float* obase = out + ((size_t)n * COUT + cout) * HWSZ + oh0 * WW2;
    #pragma unroll
    for (int mt = 0; mt < 7; ++mt) {
        const int m0 = (mhalf * 7 + mt) * 16 + mbase;    // D layout: VGPR r -> M=m0+r
        float4* o4 = (float4*)(obase + m0);
        o4[0] = make_float4(acc[mt][0], acc[mt][1], acc[mt][2], acc[mt][3]);
        o4[1] = make_float4(acc[mt][4], acc[mt][5], acc[mt][6], acc[mt][7]);
    }
}

extern "C" void kernel_launch(void* const* d_in, const int* in_sizes, int n_in,
                              void* d_out, int out_size, void* d_ws, size_t ws_size,
                              hipStream_t stream) {
    const float* x    = (const float*)d_in[0];   // (128,64,56,56) fp32
    const float* ta   = (const float*)d_in[1];   // (64,1,3) fp32
    const float* netw = (const float*)d_in[2];   // (64,64,3,3) fp32
    float* out        = (float*)d_out;           // (128,64,56,56) fp32
    __bf16* wperm     = (__bf16*)d_ws;           // 73,728 B permuted bf16 weights

    tsm_prep_weights<<<dim3((COUT * KTOT) / 256), dim3(256), 0, stream>>>(netw, wperm);

    const dim3 grid(128 * (HH / ROWS_PER_WG));   // 1792 blocks, exact cover
    tsm_conv3x3_wmma_bf16<<<grid, NTHREADS, SMEM_BYTES, stream>>>(x, ta, wperm, out);
}
